// GRU_58360015618743
// MI455X (gfx1250) — matrix operand order repaired and checked by reading, not compiled, not run on hardware
//
#include <hip/hip_runtime.h>

#define B_  256
#define T_  256
#define I_  1024
#define H_  1024
#define G_  3072   // 3*H

typedef __bf16 bf16;
typedef __attribute__((ext_vector_type(16))) __bf16 v16bf;
typedef __attribute__((ext_vector_type(8)))  __bf16 v8bf;
typedef __attribute__((ext_vector_type(8)))  float  v8f;
typedef __attribute__((ext_vector_type(4)))  float  v4f;

// ---------------- utility kernels ----------------

__global__ void cvt_bf16_kernel(const float* __restrict__ src, bf16* __restrict__ dst, int n) {
    int i = blockIdx.x * blockDim.x + threadIdx.x;
    if (i < n) dst[i] = (bf16)src[i];
}

__global__ void copy_f32_kernel(const float* __restrict__ src, float* __restrict__ dst, int n) {
    int i = blockIdx.x * blockDim.x + threadIdx.x;
    if (i < n) dst[i] = src[i];
}

// ---------------- WMMA GEMM ----------------
// C[256][3072] = A[256][K=1024] (fp32, row stride lda, optional per-row zero mask)
//              x W^T            (W bf16 [3072][1024], row-major: row = output col)
//              + bias[3072]
// Tile: BM=128 x BN=128, K-chunk 32. 256 threads = 8 wave32, wave = 64(M) x 32(N).
// B tile staged with GLOBAL_LOAD_ASYNC_TO_LDS_B128 (ASYNCcnt); A tile converted
// fp32->bf16 through VGPRs (mask folded in) and stored with ds_store_b128.

#define BM 128
#define BN 128
#define BK 32
#define LDSP 40   // padded LDS row stride (elements): 80B rows, spreads banks

static __device__ inline v16bf cat16(v8bf lo, v8bf hi) {
    v16bf r;
#pragma unroll
    for (int i = 0; i < 8; ++i) { r[i] = lo[i]; r[i + 8] = hi[i]; }
    return r;
}

// Async 16-byte global -> LDS copy (CDNA5; tracked by ASYNCcnt).
static __device__ inline void async_copy_b128(bf16* lds_dst, const bf16* gsrc) {
    unsigned int  lds_off = (unsigned int)(unsigned long long)lds_dst;  // LDS aperture offset
    unsigned long long gaddr = (unsigned long long)gsrc;
    asm volatile("global_load_async_to_lds_b128 %0, %1, off"
                 :: "v"(lds_off), "v"(gaddr)
                 : "memory");
}

static __device__ inline void wait_asynccnt0() {
    asm volatile("s_wait_asynccnt 0x0" ::: "memory");
}

__global__ __launch_bounds__(256)
void gemm_bf16_kernel(float* __restrict__ C,                 // [256][G_]
                      const float* __restrict__ A,           // row m at A + m*lda
                      long lda,
                      const bf16* __restrict__ W,            // [G_][1024] bf16
                      const float* __restrict__ bias,        // [G_]
                      const unsigned char* __restrict__ mask,// row m mask at mask[m*maskStride]; null=none
                      int maskStride)
{
    __shared__ bf16 sA[BM * LDSP];
    __shared__ bf16 sB[BN * LDSP];

    const int tid  = threadIdx.x;
    const int lane = tid & 31;
    const int wave = tid >> 5;
    const int wm   = wave & 1;    // 64-row slice of the tile
    const int wn   = wave >> 1;   // 32-col slice of the tile
    const int blockN = blockIdx.x * BN;
    const int blockM = blockIdx.y * BM;

    v8f acc[4][2];
#pragma unroll
    for (int i = 0; i < 4; ++i)
#pragma unroll
        for (int j = 0; j < 2; ++j) acc[i][j] = (v8f){0.f, 0.f, 0.f, 0.f, 0.f, 0.f, 0.f, 0.f};

    // staging: each thread owns 16 consecutive K elements of one row
    const int srow = tid >> 1;          // 0..127
    const int scol = (tid & 1) * 16;    // 0 or 16

    float rowScale = 1.0f;
    if (mask) rowScale = mask[(size_t)(blockM + srow) * maskStride] ? 0.0f : 1.0f;

    const float* aRow = A + (long)(blockM + srow) * lda;
    const bf16*  wRow = W + (size_t)(blockN + srow) * I_;
    bf16* sBdst = &sB[srow * LDSP + scol];

    // WMMA fragment addressing (ISA 7.12.2, wave32):
    // A 16x32 bf16: lanes 0-15 -> M=lane, K {0..7,16..23}; lanes 16-31 -> K {8..15,24..31}
    // B 32x16 bf16: lanes 0-15 -> N=lane, K {0..7,16..23}; lanes 16-31 -> K {8..15,24..31}
    const int fr  = lane & 15;
    const int hiA = (lane >> 4) * 8;
    const int hiB = (lane >> 4) * 16;

    for (int k0 = 0; k0 < I_; k0 += BK) {
        // A global loads issued before the barrier (overlap with prior chunk's WMMAs)
        v4f a0 = *(const v4f*)(aRow + k0 + scol + 0);
        v4f a1 = *(const v4f*)(aRow + k0 + scol + 4);
        v4f a2 = *(const v4f*)(aRow + k0 + scol + 8);
        v4f a3 = *(const v4f*)(aRow + k0 + scol + 12);

        __syncthreads();   // previous chunk's fragment reads done; LDS reusable

        // B tile: cache -> LDS without touching VGPRs (2 x 16B per thread)
        async_copy_b128(sBdst,     wRow + k0 + scol);
        async_copy_b128(sBdst + 8, wRow + k0 + scol + 8);

        // A tile: convert + mask, then ds_store
        v8bf c0, c1;
#pragma unroll
        for (int i = 0; i < 4; ++i) {
            c0[i]     = (bf16)(a0[i] * rowScale);
            c0[i + 4] = (bf16)(a1[i] * rowScale);
            c1[i]     = (bf16)(a2[i] * rowScale);
            c1[i + 4] = (bf16)(a3[i] * rowScale);
        }
        *(v8bf*)&sA[srow * LDSP + scol]     = c0;
        *(v8bf*)&sA[srow * LDSP + scol + 8] = c1;

        wait_asynccnt0();  // own async LDS writes landed (dscnt waited by compiler)
        __syncthreads();   // tile fully staged across all waves

        v16bf afrag[4], bfrag[2];
#pragma unroll
        for (int i = 0; i < 4; ++i) {
            const bf16* p = &sA[(wm * 64 + i * 16 + fr) * LDSP];
            afrag[i] = cat16(*(const v8bf*)(p + hiA), *(const v8bf*)(p + hiA + 16));
        }
#pragma unroll
        for (int j = 0; j < 2; ++j) {
            const bf16* p = &sB[(wn * 32 + j * 16 + fr) * LDSP];
            bfrag[j] = cat16(*(const v8bf*)(p + hiA), *(const v8bf*)(p + hiA + 16));
        }
#pragma unroll
        for (int i = 0; i < 4; ++i)
#pragma unroll
            for (int j = 0; j < 2; ++j)
                acc[i][j] = __builtin_amdgcn_wmma_f32_16x16x32_bf16(
                    false, afrag[i], false, bfrag[j], (short)0, acc[i][j], false, false);
    }

    // C/D layout: element (m,n): VGPR (m&7), lane n + 16*(m>=8)
    const int mBase = blockM + wm * 64 + ((lane >> 4) ? 8 : 0);
    const int nLane = lane & 15;
#pragma unroll
    for (int i = 0; i < 4; ++i) {
#pragma unroll
        for (int j = 0; j < 2; ++j) {
            const int gn = blockN + wn * 32 + j * 16 + nLane;
            const float bv = bias[gn];
            float* cp = C + (size_t)(mBase + i * 16) * G_ + gn;
#pragma unroll
            for (int r = 0; r < 8; ++r)
                cp[(size_t)r * G_] = acc[i][j][r] + bv;
        }
    }
}

// ---------------- fused GRU gate + output ----------------

__global__ __launch_bounds__(256)
void gru_gate_kernel(float* __restrict__ hbuf,
                     float* __restrict__ out,                 // output part 1 (x + h_t)
                     const float* __restrict__ hp,            // [256][3072] = mask(h) @ W_hh^T + b_hh
                     const float* __restrict__ xp,            // [256][3072] = x_t @ W_ih^T + b_ih
                     const float* __restrict__ x,
                     const unsigned char* __restrict__ is_init,
                     int t)
{
    const int idx = blockIdx.x * 256 + threadIdx.x;  // over B*H
    const int b = idx >> 10;
    const int j = idx & 1023;

    float h = hbuf[idx];
    if (is_init[b * T_ + t]) h = 0.0f;

    const float* hpb = hp + (size_t)b * G_;
    const float* xpb = xp + (size_t)b * G_;
    const float pr = xpb[j]          + hpb[j];
    const float pz = xpb[j + H_]     + hpb[j + H_];
    const float xn = xpb[j + 2 * H_];
    const float hn = hpb[j + 2 * H_];

    const float r = 1.0f / (1.0f + __expf(-pr));
    const float z = 1.0f / (1.0f + __expf(-pz));
    const float n = tanhf(xn + r * hn);
    const float hnew = (1.0f - z) * n + z * h;

    hbuf[idx] = hnew;
    const size_t o = ((size_t)b * T_ + t) * H_ + j;
    out[o] = x[o] + hnew;
}

// ---------------- broadcast h_last ----------------

__global__ __launch_bounds__(256)
void bcast_kernel(float* __restrict__ out2, const float* __restrict__ hbuf)
{
    const size_t idx = (size_t)blockIdx.x * 256 + threadIdx.x;  // over B*T*H
    const int j  = (int)(idx & 1023);
    const int b  = (int)(idx >> 18);                            // / (T_*H_)
    out2[idx] = hbuf[(size_t)b * H_ + j];
}

// ---------------- launcher ----------------

extern "C" void kernel_launch(void* const* d_in, const int* in_sizes, int n_in,
                              void* d_out, int out_size, void* d_ws, size_t ws_size,
                              hipStream_t stream)
{
    (void)in_sizes; (void)n_in; (void)out_size; (void)ws_size;

    const float*         x       = (const float*)d_in[0];
    const unsigned char* is_init = (const unsigned char*)d_in[1];   // jnp.bool_ = 1 byte
    const float*         hx      = (const float*)d_in[2];
    const float*         W_ih    = (const float*)d_in[3];
    const float*         W_hh    = (const float*)d_in[4];
    const float*         b_ih    = (const float*)d_in[5];
    const float*         b_hh    = (const float*)d_in[6];
    float* out = (float*)d_out;

    char* ws = (char*)d_ws;
    bf16* Wih_bf = (bf16*)ws;  ws += (size_t)G_ * I_ * sizeof(bf16);
    bf16* Whh_bf = (bf16*)ws;  ws += (size_t)G_ * H_ * sizeof(bf16);
    float* hp    = (float*)ws; ws += (size_t)B_ * G_ * sizeof(float);
    float* xp    = (float*)ws; ws += (size_t)B_ * G_ * sizeof(float);
    float* hbuf  = (float*)ws;

    const int nW = G_ * I_;
    cvt_bf16_kernel<<<(nW + 255) / 256, 256, 0, stream>>>(W_ih, Wih_bf, nW);
    cvt_bf16_kernel<<<(nW + 255) / 256, 256, 0, stream>>>(W_hh, Whh_bf, nW);
    copy_f32_kernel<<<(B_ * H_ + 255) / 256, 256, 0, stream>>>(hx, hbuf, B_ * H_);

    const dim3 ggrid(G_ / BN, B_ / BM);   // (24, 2)
    for (int t = 0; t < T_; ++t) {
        // hp = mask(h) @ W_hh^T + b_hh   (mask folded into A staging)
        gemm_bf16_kernel<<<ggrid, 256, 0, stream>>>(
            hp, hbuf, (long)H_, Whh_bf, b_hh, is_init + t, T_);
        // xp = x_t @ W_ih^T + b_ih       (row b of A is x[b, t, :], stride T*I)
        gemm_bf16_kernel<<<ggrid, 256, 0, stream>>>(
            xp, x + (size_t)t * I_, (long)T_ * I_, Wih_bf, b_ih, nullptr, 0);
        // gates + state update + output
        gru_gate_kernel<<<(B_ * H_) / 256, 256, 0, stream>>>(
            hbuf, out, hp, xp, x, is_init, t);
    }

    float* out2 = out + (size_t)B_ * T_ * H_;
    bcast_kernel<<<(unsigned)(((size_t)B_ * T_ * H_) / 256), 256, 0, stream>>>(out2, hbuf);
}
